// MHA_18983755448543
// MI455X (gfx1250) — compile-verified, matrix operand order checked by reading, not run on hardware
//
#include <hip/hip_runtime.h>
#include <hip/hip_bf16.h>
#include <hip/hip_fp16.h>

// ---------------------------------------------------------------------------
// MHA forward for MI455X (gfx1250): f16 WMMA + Tensor Data Mover (TDM)
// double-buffered GEMM tiles.
// ---------------------------------------------------------------------------

typedef __attribute__((ext_vector_type(4)))  _Float16 v4h_t;
typedef __attribute__((ext_vector_type(8)))  _Float16 v8h_t;
typedef __attribute__((ext_vector_type(16))) _Float16 v16h_t;
typedef __attribute__((ext_vector_type(8)))  float    v8f_t;

#define NUM_HEADS 16
#define EMBED     1024
#define HEAD_DIM  64
#define BATCH     4
#define SEQ       2048
#define MTOKENS   (BATCH * SEQ)   // 8192

#ifndef __has_builtin
#define __has_builtin(x) 0
#endif
#if __has_builtin(__builtin_amdgcn_tensor_load_to_lds) && \
    __has_builtin(__builtin_amdgcn_s_wait_tensorcnt)
#define USE_TDM 1
#else
#define USE_TDM 0
#endif

union AFrag { v16h_t v; v8h_t h[2]; };
union Acc8  { v8f_t  v; float  f[8]; };

static __device__ __forceinline__ v8f_t wmma_f16(v16h_t a, v16h_t b, v8f_t c) {
  // (neg_a, A, neg_b, B, c_mod, C, reuse_a, reuse_b)
  return __builtin_amdgcn_wmma_f32_16x16x32_f16(false, a, false, b, (short)0, c,
                                                false, false);
}

#if USE_TDM
typedef __attribute__((ext_vector_type(4))) unsigned int u32x4_t;
typedef __attribute__((ext_vector_type(8))) int          i32x8_t;
typedef __attribute__((ext_vector_type(4))) int          i32x4_t;

// 2D f16 tile DMA: global (row-major, row stride `stride_e` elements) -> LDS.
// D# built per CDNA5 ISA; tile fully in-bounds so
// tensor_dim == tile_dim; data_size=1 (2 bytes); groups 2/3 unused (2D).
// This toolchain's builtin is the 6-arg clang-23 form:
//   (uint32x4 g0, int32x8 g1, int32x4, int32x4, int32x8, i32 cpol)
static __device__ __forceinline__ void tdm_load_tile_f16(
    unsigned lds_byte, const _Float16* gptr, unsigned rows, unsigned cols,
    unsigned stride_e) {
  unsigned long long ga = (unsigned long long)(size_t)gptr;
  u32x4_t g0;
  g0[0] = 1u;                                                // count=1
  g0[1] = lds_byte;                                          // lds_addr
  g0[2] = (unsigned)ga;                                      // global_addr lo
  g0[3] = (unsigned)((ga >> 32) & 0x01FFFFFFull) | (2u << 30); // hi | type=2
  i32x8_t g1;
  g1[0] = (int)(1u << 16);                                   // data_size=2B
  g1[1] = (int)((cols & 0xFFFFu) << 16);                     // tensor_dim0 lo
  g1[2] = (int)(((cols >> 16) & 0xFFFFu) | ((rows & 0xFFFFu) << 16));
  g1[3] = (int)(((rows >> 16) & 0xFFFFu) | ((cols & 0xFFFFu) << 16)); // tile_dim0
  g1[4] = (int)(rows & 0xFFFFu);                             // tile_dim1
  g1[5] = (int)stride_e;                                     // dim0_stride lo
  g1[6] = 0;
  g1[7] = 0;
  i32x4_t z4 = {};
  i32x8_t z8 = {};
  __builtin_amdgcn_tensor_load_to_lds(g0, g1, z4, z4, z8, 0);
}
#endif

// ---------------------------------------------------------------------------
// 1) Embedding gather + f32 -> f16 cast.  One thread = 4 elements.
// ---------------------------------------------------------------------------
__global__ __launch_bounds__(256) void k_embed_cast(
    const int* __restrict__ inp, const float* __restrict__ emb,
    _Float16* __restrict__ xh) {
  size_t t = (size_t)blockIdx.x * blockDim.x + threadIdx.x;
  int row  = (int)(t >> 8);            // EMBED/4 = 256 vec4 per row
  int col4 = (int)(t & 255) * 4;
  int tok  = inp[row];
  const float4 f = *(const float4*)&emb[(size_t)tok * EMBED + col4];
  v4h_t o = { (_Float16)f.x, (_Float16)f.y, (_Float16)f.z, (_Float16)f.w };
  *(v4h_t*)&xh[(size_t)row * EMBED + col4] = o;
}

// ---------------------------------------------------------------------------
// 2) Plain f32 -> f16 cast (weights).  One thread = 4 elements.
// ---------------------------------------------------------------------------
__global__ __launch_bounds__(256) void k_cast_f32_f16(
    const float* __restrict__ src, _Float16* __restrict__ dst, int n4) {
  int t = blockIdx.x * 256 + threadIdx.x;
  if (t >= n4) return;
  const float4 f = *(const float4*)&src[(size_t)t * 4];
  v4h_t o = { (_Float16)f.x, (_Float16)f.y, (_Float16)f.z, (_Float16)f.w };
  *(v4h_t*)&dst[(size_t)t * 4] = o;
}

// ---------------------------------------------------------------------------
// 3) GEMM: C[M,N] = A[M,K] (f16) * B[N,K]^T (f16) + bias, C in f32 or f16.
//    Block tile 64x128, 8 waves (2M x 4N), wave tile 32x32 (2x2 WMMA),
//    K step 32.  Tiles move via TDM (double-buffered, TENSORcnt-pipelined)
//    when the builtin is available; manual global->LDS copies otherwise.
// ---------------------------------------------------------------------------
__global__ __launch_bounds__(256) void k_gemm_xwt(
    const _Float16* __restrict__ A, const _Float16* __restrict__ Bw,
    const float* __restrict__ bias, float* __restrict__ Cf,
    _Float16* __restrict__ Ch, int M, int N, int K) {
#if USE_TDM
  __shared__ __align__(16) _Float16 sA[2][64 * 32];
  __shared__ __align__(16) _Float16 sB[2][128 * 32];
#else
  __shared__ __align__(16) _Float16 sA[1][64 * 32];
  __shared__ __align__(16) _Float16 sB[1][128 * 32];
#endif

  const int tid  = threadIdx.x;
  const int lane = tid & 31;
  const int wave = tid >> 5;
  const int bm   = blockIdx.x * 64;
  const int bn   = blockIdx.y * 128;
  const int wm   = (wave & 1) * 32;
  const int wn   = (wave >> 1) * 32;
  const int lr   = lane & 15;          // fragment row (A) / column (B)
  const int lc   = (lane >> 4) * 8;    // fragment k-chunk base

  Acc8 acc[2][2];
  { v8f_t z = {}; for (int i = 0; i < 2; ++i) for (int j = 0; j < 2; ++j) acc[i][j].v = z; }

#if USE_TDM
  const unsigned ldsA0 = (unsigned)(size_t)&sA[0][0];
  const unsigned ldsA1 = (unsigned)(size_t)&sA[1][0];
  const unsigned ldsB0 = (unsigned)(size_t)&sB[0][0];
  const unsigned ldsB1 = (unsigned)(size_t)&sB[1][0];
  if (wave == 0) {  // prologue DMA: tile 0 -> buffer 0
    tdm_load_tile_f16(ldsA0, &A [(size_t)bm * K], 64, 32, (unsigned)K);
    tdm_load_tile_f16(ldsB0, &Bw[(size_t)bn * K], 128, 32, (unsigned)K);
  }
#else
  const int ldr = tid >> 2;            // LDS fill row
  const int ldc = (tid & 3) * 8;       // LDS fill col
#endif

  int buf = 0;
  for (int k0 = 0; k0 < K; k0 += 32) {
#if USE_TDM
    if (wave == 0) {
      if (k0 + 32 < K) {  // kick next tile into the other buffer, then wait
        tdm_load_tile_f16(buf ? ldsA0 : ldsA1,
                          &A [(size_t)bm * K + k0 + 32], 64, 32, (unsigned)K);
        tdm_load_tile_f16(buf ? ldsB0 : ldsB1,
                          &Bw[(size_t)bn * K + k0 + 32], 128, 32, (unsigned)K);
        __builtin_amdgcn_s_wait_tensorcnt(2);  // current tile landed
      } else {
        __builtin_amdgcn_s_wait_tensorcnt(0);
      }
    }
    __syncthreads();
#else
    *(v8h_t*)&sA[0][ldr * 32 + ldc]        = *(const v8h_t*)&A [(size_t)(bm + ldr)      * K + k0 + ldc];
    *(v8h_t*)&sB[0][ldr * 32 + ldc]        = *(const v8h_t*)&Bw[(size_t)(bn + ldr)      * K + k0 + ldc];
    *(v8h_t*)&sB[0][(ldr + 64) * 32 + ldc] = *(const v8h_t*)&Bw[(size_t)(bn + ldr + 64) * K + k0 + ldc];
    if (k0 + 32 < K) {
      __builtin_prefetch(&A [(size_t)(bm + ldr) * K + k0 + 32 + ldc], 0, 1);
      __builtin_prefetch(&Bw[(size_t)(bn + ldr) * K + k0 + 32 + ldc], 0, 1);
    }
    __syncthreads();
#endif

    AFrag bfr[2];
    for (int j = 0; j < 2; ++j) {
      bfr[j].h[0] = *(const v8h_t*)&sB[buf][(wn + j * 16 + lr) * 32 + lc];
      bfr[j].h[1] = *(const v8h_t*)&sB[buf][(wn + j * 16 + lr) * 32 + lc + 16];
    }
    for (int i = 0; i < 2; ++i) {
      AFrag af;
      af.h[0] = *(const v8h_t*)&sA[buf][(wm + i * 16 + lr) * 32 + lc];
      af.h[1] = *(const v8h_t*)&sA[buf][(wm + i * 16 + lr) * 32 + lc + 16];
      for (int j = 0; j < 2; ++j)
        acc[i][j].v = wmma_f16(af.v, bfr[j].v, acc[i][j].v);
    }
    __syncthreads();   // buffer drained before the DMA that overwrites it
#if USE_TDM
    buf ^= 1;
#endif
  }

  // C/D layout: element e of v8f -> row e + (lane>>4)*8, col lane&15.
  const int rhi = (lane >> 4) * 8;
  for (int i = 0; i < 2; ++i)
    for (int j = 0; j < 2; ++j) {
      const int col  = bn + wn + j * 16 + lr;
      const float bv = bias[col];
      for (int e = 0; e < 8; ++e) {
        const size_t row = (size_t)(bm + wm + i * 16 + rhi + e);
        const float val  = acc[i][j].f[e] + bv;
        if (Cf) Cf[row * N + col] = val;
        else    Ch[row * N + col] = (_Float16)val;
      }
    }
}

// ---------------------------------------------------------------------------
// 4) V transpose: vh[(b*S+s)*E + h*64 + d] -> vt[((b*H+h)*64 + d)*S + s]
// ---------------------------------------------------------------------------
__global__ __launch_bounds__(256) void k_transpose_v(
    const _Float16* __restrict__ vh, _Float16* __restrict__ vt) {
  size_t t = (size_t)blockIdx.x * 256 + threadIdx.x;    // 0 .. M*E-1
  int s    = (int)(t & (SEQ - 1));
  int rest = (int)(t >> 11);                            // /SEQ
  int d    = rest & (HEAD_DIM - 1);
  int h    = (rest >> 6) & (NUM_HEADS - 1);
  int b    = rest >> 10;                                // /(64*16)
  vt[t] = vh[(size_t)(b * SEQ + s) * EMBED + h * HEAD_DIM + d];
}

// ---------------------------------------------------------------------------
// 5) Flash attention.  Each wave owns 16 query rows; per 32-key step:
//    4 WMMA (QK^T) + online softmax (shfl over 16-lane groups) + P staged
//    through wave-private LDS (C-layout -> A-layout) + 4 WMMA (P*V^T).
// ---------------------------------------------------------------------------
__global__ __launch_bounds__(256) void k_flash_attn(
    const _Float16* __restrict__ Qm, const _Float16* __restrict__ Km,
    const _Float16* __restrict__ Vt, _Float16* __restrict__ Om) {
  __shared__ __align__(16) _Float16 sP[8][16 * 32];

  const int lane = threadIdx.x & 31;
  const int wave = threadIdx.x >> 5;
  const int lr   = lane & 15;
  const int lc   = (lane >> 4) * 8;
  const int bh   = blockIdx.y;                 // 0 .. B*H-1
  const int b    = bh >> 4;
  const int h    = bh & (NUM_HEADS - 1);
  const int q0   = blockIdx.x * 128 + wave * 16;

  const _Float16* qp = Qm + (size_t)(b * SEQ + q0) * EMBED + h * HEAD_DIM;
  const _Float16* kp = Km + (size_t)(b * SEQ) * EMBED + h * HEAD_DIM;
  const _Float16* vp = Vt + (size_t)bh * HEAD_DIM * SEQ;

  AFrag qf[2];                                  // Q 16x64 = two K=32 A-frags
  for (int kk = 0; kk < 2; ++kk) {
    qf[kk].h[0] = *(const v8h_t*)&qp[(size_t)lr * EMBED + kk * 32 + lc];
    qf[kk].h[1] = *(const v8h_t*)&qp[(size_t)lr * EMBED + kk * 32 + lc + 16];
  }

  Acc8 oacc[4];
  { v8f_t z = {}; for (int c = 0; c < 4; ++c) oacc[c].v = z; }
  float mrun[8], lrun[8];
  for (int e = 0; e < 8; ++e) { mrun[e] = -1e30f; lrun[e] = 0.0f; }

  const float scale = 0.125f;                   // 1/sqrt(64)

  for (int j0 = 0; j0 < SEQ; j0 += 32) {
    Acc8 sc[2];
    for (int t = 0; t < 2; ++t) {
      const _Float16* kr = &kp[(size_t)(j0 + t * 16 + lr) * EMBED];
      AFrag kf0, kf1;
      kf0.h[0] = *(const v8h_t*)&kr[lc];
      kf0.h[1] = *(const v8h_t*)&kr[lc + 16];
      kf1.h[0] = *(const v8h_t*)&kr[32 + lc];
      kf1.h[1] = *(const v8h_t*)&kr[32 + lc + 16];
      v8f_t z = {};
      z = wmma_f16(qf[0].v, kf0.v, z);
      z = wmma_f16(qf[1].v, kf1.v, z);
      sc[t].v = z;
    }

    for (int e = 0; e < 8; ++e) {
      float s0 = sc[0].f[e] * scale;
      float s1 = sc[1].f[e] * scale;
      float mx = fmaxf(s0, s1);
      for (int off = 8; off >= 1; off >>= 1)
        mx = fmaxf(mx, __shfl_xor(mx, off, 16));
      const float mnew = fmaxf(mrun[e], mx);
      const float p0 = __expf(s0 - mnew);
      const float p1 = __expf(s1 - mnew);
      float ps = p0 + p1;
      for (int off = 8; off >= 1; off >>= 1)
        ps += __shfl_xor(ps, off, 16);
      const float corr = __expf(mrun[e] - mnew);
      lrun[e] = lrun[e] * corr + ps;
      mrun[e] = mnew;
      const int prow = e + (lane >> 4) * 8;
      sP[wave][prow * 32 + lr]      = (_Float16)p0;   // C-layout -> LDS
      sP[wave][prow * 32 + 16 + lr] = (_Float16)p1;
      for (int c = 0; c < 4; ++c) oacc[c].f[e] *= corr;
    }

    AFrag pf;  // P (16x32) back as one A-fragment (in-order DS, same wave)
    pf.h[0] = *(const v8h_t*)&sP[wave][lr * 32 + lc];
    pf.h[1] = *(const v8h_t*)&sP[wave][lr * 32 + lc + 16];

    for (int c = 0; c < 4; ++c) {
      const _Float16* vr = &vp[(size_t)(c * 16 + lr) * SEQ + j0];
      AFrag vf;
      vf.h[0] = *(const v8h_t*)&vr[lc];
      vf.h[1] = *(const v8h_t*)&vr[lc + 16];
      oacc[c].v = wmma_f16(pf.v, vf.v, oacc[c].v);
    }
  }

  for (int e = 0; e < 8; ++e) {
    const float inv  = 1.0f / lrun[e];
    const size_t row = (size_t)(b * SEQ + q0 + e + (lane >> 4) * 8);
    for (int c = 0; c < 4; ++c)
      Om[row * EMBED + h * HEAD_DIM + c * 16 + lr] =
          (_Float16)(oacc[c].f[e] * inv);
  }
}

// ---------------------------------------------------------------------------
// Host side
// ---------------------------------------------------------------------------
extern "C" void kernel_launch(void* const* d_in, const int* in_sizes, int n_in,
                              void* d_out, int out_size, void* d_ws,
                              size_t ws_size, hipStream_t stream) {
  (void)in_sizes; (void)n_in; (void)out_size; (void)ws_size;

  const int*   inp = (const int*)  d_in[0];
  const float* emb = (const float*)d_in[1];
  const float* Wq  = (const float*)d_in[2];
  const float* bq  = (const float*)d_in[3];
  const float* Wk  = (const float*)d_in[4];
  const float* bk  = (const float*)d_in[5];
  const float* Wv  = (const float*)d_in[6];
  const float* bv  = (const float*)d_in[7];
  const float* Wo  = (const float*)d_in[8];
  const float* bo  = (const float*)d_in[9];
  float* out = (float*)d_out;

  const size_t ME = (size_t)MTOKENS * EMBED;   // 8,388,608 halves
  const size_t EE = (size_t)EMBED * EMBED;     // 1,048,576 halves

  _Float16* p   = (_Float16*)d_ws;
  _Float16* xh  = p; p += ME;
  _Float16* wqh = p; p += EE;
  _Float16* wkh = p; p += EE;
  _Float16* wvh = p; p += EE;
  _Float16* woh = p; p += EE;
  _Float16* qh  = p; p += ME;
  _Float16* kh  = p; p += ME;
  _Float16* vh  = p; p += ME;
  _Float16* vt  = p; p += ME;
  _Float16* ctx = p; p += ME;

  k_embed_cast<<<(unsigned)(ME / 4 / 256), 256, 0, stream>>>(inp, emb, xh);
  const int n4 = (int)(EE / 4);
  k_cast_f32_f16<<<(unsigned)(EE / 4 / 256), 256, 0, stream>>>(Wq, wqh, n4);
  k_cast_f32_f16<<<(unsigned)(EE / 4 / 256), 256, 0, stream>>>(Wk, wkh, n4);
  k_cast_f32_f16<<<(unsigned)(EE / 4 / 256), 256, 0, stream>>>(Wv, wvh, n4);
  k_cast_f32_f16<<<(unsigned)(EE / 4 / 256), 256, 0, stream>>>(Wo, woh, n4);

  dim3 gg(MTOKENS / 64, EMBED / 128);
  k_gemm_xwt<<<gg, 256, 0, stream>>>(xh, wqh, bq, nullptr, qh, MTOKENS, EMBED, EMBED);
  k_gemm_xwt<<<gg, 256, 0, stream>>>(xh, wkh, bk, nullptr, kh, MTOKENS, EMBED, EMBED);
  k_gemm_xwt<<<gg, 256, 0, stream>>>(xh, wvh, bv, nullptr, vh, MTOKENS, EMBED, EMBED);

  k_transpose_v<<<(unsigned)(ME / 256), 256, 0, stream>>>(vh, vt);

  dim3 ga(SEQ / 128, BATCH * NUM_HEADS);
  k_flash_attn<<<ga, 256, 0, stream>>>(qh, kh, vt, ctx);

  k_gemm_xwt<<<gg, 256, 0, stream>>>(ctx, woh, bo, out, nullptr, MTOKENS, EMBED, EMBED);
}
